// CRFOutputLayer_85856396247806
// MI455X (gfx1250) — matrix operand order repaired and checked by reading, not compiled, not run on hardware
//
#include <hip/hip_runtime.h>
#include <hip/hip_bf16.h>
#include <math.h>

#define Bn 128
#define Tn 512
#define Kn 256

typedef _Float16 v16h __attribute__((ext_vector_type(16)));
typedef float    v8f  __attribute__((ext_vector_type(8)));

#define NEG_BIG (-3.402823466e38f)

__device__ __forceinline__ float waveMax(float v) {
#pragma unroll
  for (int m = 16; m >= 1; m >>= 1) v = fmaxf(v, __shfl_xor(v, m, 32));
  return v;
}
__device__ __forceinline__ float waveSum(float v) {
#pragma unroll
  for (int m = 16; m >= 1; m >>= 1) v += __shfl_xor(v, m, 32);
  return v;
}
// max with first-index tie-break (matches jnp.argmax)
__device__ __forceinline__ void waveArgmax(float& v, int& idx) {
#pragma unroll
  for (int m = 16; m >= 1; m >>= 1) {
    float ov = __shfl_xor(v, m, 32);
    int   oi = __shfl_xor(idx, m, 32);
    if (ov > v || (ov == v && oi < idx)) { v = ov; idx = oi; }
  }
}

// ---------------------------------------------------------------------------
// Kernel 1: R = expm1(transitions) packed as f16 in WMMA B-fragment order.
// B fragment (32K x 16N, f16): lane = n + 16*(k>=16), half slot = k & 15.
// Fragment order: (ntile * 8 + ktile) * 512 halves.
// ---------------------------------------------------------------------------
__global__ __launch_bounds__(256) void prep_rfrag_kernel(
    const float* __restrict__ trans, _Float16* __restrict__ Rfrag) {
  int idx = blockIdx.x * 256 + threadIdx.x;  // idx = i*Kn + j
  int i = idx >> 8, j = idx & 255;
  float r = expm1f(trans[idx]);
  int kt = i >> 5, ki = i & 31;
  int nt = j >> 4, ni = j & 15;
  int lane = ni + 16 * (ki >> 4);
  int h = ki & 15;
  Rfrag[(size_t)((nt * 8 + kt) << 9) + lane * 16 + h] = (_Float16)r;
}

// ---------------------------------------------------------------------------
// Kernel 2: numerator path score per batch (one wave per batch)
// ---------------------------------------------------------------------------
__global__ __launch_bounds__(32) void score_kernel(
    const float* __restrict__ logit, const float* __restrict__ trans,
    const float* __restrict__ startT, const float* __restrict__ endT,
    const int* __restrict__ target, const int* __restrict__ seq_lens,
    float* __restrict__ score) {
  int b = blockIdx.x;
  int lane = threadIdx.x;
  int len = seq_lens[b];
  float acc = 0.f;
  for (int t = 1 + lane; t < Tn; t += 32) {
    if (t < len) {
      int tp = target[b * Tn + t - 1];
      int tc = target[b * Tn + t];
      acc += trans[tp * Kn + tc] + logit[((size_t)b * Tn + t) * Kn + tc];
    }
  }
  acc = waveSum(acc);
  if (lane == 0) {
    int t0 = target[b * Tn];
    acc += startT[t0] + logit[(size_t)b * Tn * Kn + t0];
    int tl = target[b * Tn + len - 1];
    acc += endT[tl];
    score[b] = acc;
  }
}

// ---------------------------------------------------------------------------
// Kernel 3: forward scan via WMMA f16 GEMM (16 batch rows / block, 16 waves,
// wave w owns output N-tile w, its 8 R B-fragments stay in registers all T).
// ---------------------------------------------------------------------------
__global__ __launch_bounds__(512) void fwd_kernel(
    const float* __restrict__ logit, const float* __restrict__ startT,
    const float* __restrict__ endT, const int* __restrict__ seq_lens,
    const _Float16* __restrict__ Rfrag, float* __restrict__ logZ) {
  __shared__ float alpha[16 * 256];      // 16 batch rows x K
  __shared__ v16h  PfragV[8 * 32];       // A fragments: 8 ktiles x 32 lanes
  __shared__ float mRow[16], sRow[16];
  __shared__ int   slen[16];
  _Float16* Pfrag = (_Float16*)PfragV;

  const int tid = threadIdx.x;
  const int wave = tid >> 5;
  const int lane = tid & 31;
  const int b0 = blockIdx.x * 16;

  if (tid < 16) slen[tid] = seq_lens[b0 + tid];

  // Resident B fragments of R for this wave's N-tile
  v16h breg[8];
#pragma unroll
  for (int kt = 0; kt < 8; ++kt)
    breg[kt] = *(const v16h*)(Rfrag + (size_t)((wave * 8 + kt) << 9) + lane * 16);

  // alpha0 = start + emissions[:,0]
  for (int x = tid; x < 16 * 256; x += 512) {
    int r = x >> 8, k = x & 255;
    alpha[x] = startT[k] + logit[((size_t)(b0 + r) * Tn) * Kn + k];
  }
  __syncthreads();

  for (int t = 1; t < Tn; ++t) {
    // Phase 1: wave w computes row r=w: m, S, P=exp(alpha-m) into A-fragments
    {
      const int r = wave;
      float a[8];
      float mv = NEG_BIG;
#pragma unroll
      for (int q = 0; q < 8; ++q) {
        a[q] = alpha[r * 256 + lane + 32 * q];
        mv = fmaxf(mv, a[q]);
      }
      mv = waveMax(mv);
      // A-fragment placement for k-within-tile = lane:
      int hi = lane >> 4;
      int h  = lane & 15;
      float sv = 0.f;
#pragma unroll
      for (int q = 0; q < 8; ++q) {
        float p = expf(a[q] - mv);
        sv += p;
        Pfrag[q * 512 + (r + 16 * hi) * 16 + h] = (_Float16)p;
      }
      sv = waveSum(sv);
      if (lane == 0) { mRow[r] = mv; sRow[r] = sv; }
    }
    __syncthreads();

    // Phase 2: wave w computes output tile nt=w: C = P @ R, alpha update
    {
      const int nt = wave;
      v8f c = {};
#pragma unroll
      for (int kt = 0; kt < 8; ++kt) {
        v16h afrag = PfragV[kt * 32 + lane];
        c = __builtin_amdgcn_wmma_f32_16x16x32_f16(
            false, afrag, false, breg[kt], (short)0, c, false, false);
      }
      const int N = lane & 15;
      const int mOff = (lane >> 4) * 8;
      const int col = nt * 16 + N;
#pragma unroll
      for (int v = 0; v < 8; ++v) {
        int M = v + mOff;
        float y = sRow[M] + c[v];                       // sum_i exp(a_i - m)*exp(T_ij)
        float na = mRow[M] + logf(fmaxf(y, 1e-30f)) +
                   logit[((size_t)(b0 + M) * Tn + t) * Kn + col];
        if (t < slen[M]) alpha[M * 256 + col] = na;     // jnp.where(mask, new, alpha)
      }
    }
    __syncthreads();
  }

  // logZ = logsumexp(alphaT + end)
  {
    const int r = wave;
    float a[8];
    float mv = NEG_BIG;
#pragma unroll
    for (int q = 0; q < 8; ++q) {
      int k = lane + 32 * q;
      a[q] = alpha[r * 256 + k] + endT[k];
      mv = fmaxf(mv, a[q]);
    }
    mv = waveMax(mv);
    float sv = 0.f;
#pragma unroll
    for (int q = 0; q < 8; ++q) sv += expf(a[q] - mv);
    sv = waveSum(sv);
    if (lane == 0) logZ[b0 + r] = mv + logf(sv);
  }
}

// ---------------------------------------------------------------------------
// Kernel 4: loss = mean(logZ - score)
// ---------------------------------------------------------------------------
__global__ __launch_bounds__(32) void loss_kernel(
    const float* __restrict__ logZ, const float* __restrict__ score,
    float* __restrict__ out_loss) {
  float acc = 0.f;
  for (int b = threadIdx.x; b < Bn; b += 32) acc += logZ[b] - score[b];
  acc = waveSum(acc);
  if (threadIdx.x == 0) out_loss[0] = acc / (float)Bn;
}

// ---------------------------------------------------------------------------
// Kernel 5: Viterbi max-plus scan, one batch per block (256 threads = cols),
// transitions staged through LDS 32 rows at a time; bp stored as u8.
// ---------------------------------------------------------------------------
__global__ __launch_bounds__(256) void viterbi_kernel(
    const float* __restrict__ logit, const float* __restrict__ trans,
    const float* __restrict__ startT, const float* __restrict__ endT,
    const int* __restrict__ seq_lens, unsigned char* __restrict__ bp,
    int* __restrict__ best_last) {
  __shared__ float trTile[32 * 256];  // 32 KB
  __shared__ float aCur[256];
  __shared__ float aNew[256];
  __shared__ float redV[256];
  __shared__ int   redI[256];

  const int b = blockIdx.x;
  const int tid = threadIdx.x;
  const int len = seq_lens[b];

  aCur[tid] = startT[tid] + logit[(size_t)b * Tn * Kn + tid];
  __syncthreads();

  for (int t = 1; t < Tn; ++t) {
    if (t < len) {  // block-uniform branch
      float best = NEG_BIG;
      int bi = 0;
      for (int i0 = 0; i0 < Kn; i0 += 32) {
        for (int x = tid; x < 32 * 256; x += 256)
          trTile[x] = trans[(i0 + (x >> 8)) * Kn + (x & 255)];
        __syncthreads();
#pragma unroll
        for (int ii = 0; ii < 32; ++ii) {
          float s = aCur[i0 + ii] + trTile[ii * 256 + tid];
          if (s > best) { best = s; bi = i0 + ii; }  // first-max on ties
        }
        __syncthreads();
      }
      aNew[tid] = best + logit[((size_t)b * Tn + t) * Kn + tid];
      bp[((size_t)b * Tn + t) * Kn + tid] = (unsigned char)bi;
      __syncthreads();
      aCur[tid] = aNew[tid];
      __syncthreads();
    } else {
      bp[((size_t)b * Tn + t) * Kn + tid] = (unsigned char)tid;  // identity
    }
  }

  // best_last = argmax(alphaT + end), first index on ties
  redV[tid] = aCur[tid] + endT[tid];
  redI[tid] = tid;
  __syncthreads();
  for (int s = 128; s >= 1; s >>= 1) {
    if (tid < s) {
      float ov = redV[tid + s]; int oi = redI[tid + s];
      if (ov > redV[tid] || (ov == redV[tid] && oi < redI[tid])) {
        redV[tid] = ov; redI[tid] = oi;
      }
    }
    __syncthreads();
  }
  if (tid == 0) best_last[b] = redI[0];
}

// ---------------------------------------------------------------------------
// Kernel 6: backtrack (one thread per batch)
// ---------------------------------------------------------------------------
__global__ __launch_bounds__(128) void backtrack_kernel(
    const unsigned char* __restrict__ bp, const int* __restrict__ best_last,
    int* __restrict__ pred_i, float* __restrict__ pred_f) {
  int b = threadIdx.x;
  int tag = best_last[b];
  pred_i[b * Tn + Tn - 1] = tag;
  pred_f[b * Tn + Tn - 1] = (float)tag;
  for (int t = Tn - 1; t >= 1; --t) {
    tag = bp[((size_t)b * Tn + t) * Kn + tag];
    pred_i[b * Tn + t - 1] = tag;
    pred_f[b * Tn + t - 1] = (float)tag;
  }
}

// ---------------------------------------------------------------------------
// Kernel 7: rearrange (swap pred<->argmax values) + log_softmax.
// The swap preserves the value multiset, so lse(out) == lse(logit).
// One wave per (b,t).
// ---------------------------------------------------------------------------
__global__ __launch_bounds__(256) void out_kernel(
    const float* __restrict__ logit, const int* __restrict__ pred_i,
    float* __restrict__ out_lp) {
  const int wave = threadIdx.x >> 5;
  const int lane = threadIdx.x & 31;
  const size_t bt = (size_t)blockIdx.x * 8 + wave;
  const float* x = logit + bt * Kn;

  float v[8];
  float mv = NEG_BIG;
  int mi = 0;
#pragma unroll
  for (int q = 0; q < 8; ++q) {
    int k = lane + 32 * q;
    v[q] = x[k];
    if (v[q] > mv) { mv = v[q]; mi = k; }  // ascending k => first-max per lane
  }
  waveArgmax(mv, mi);
  float sum = 0.f;
#pragma unroll
  for (int q = 0; q < 8; ++q) sum += expf(v[q] - mv);
  sum = waveSum(sum);
  const float lse = mv + logf(sum);

  const int p = pred_i[bt];
  const float labv = x[p];   // broadcast load
  const float maxv = mv;
#pragma unroll
  for (int q = 0; q < 8; ++q) {
    int k = lane + 32 * q;
    float o = (k == p) ? maxv : ((k == mi) ? labv : v[q]);
    out_lp[bt * Kn + k] = o - lse;
  }
}

// ---------------------------------------------------------------------------
extern "C" void kernel_launch(void* const* d_in, const int* in_sizes, int n_in,
                              void* d_out, int out_size, void* d_ws, size_t ws_size,
                              hipStream_t stream) {
  const float* logit    = (const float*)d_in[0];
  const float* trans    = (const float*)d_in[1];
  const float* startT   = (const float*)d_in[2];
  const float* endT     = (const float*)d_in[3];
  const int*   target   = (const int*)d_in[4];
  const int*   seq_lens = (const int*)d_in[5];

  char* ws = (char*)d_ws;
  _Float16*      Rfrag     = (_Float16*)(ws);            // 131072 B
  float*         score     = (float*)(ws + 131072);      // 512 B
  float*         logZ      = (float*)(ws + 131584);      // 512 B
  int*           best_last = (int*)(ws + 132096);        // 512 B
  int*           pred_i    = (int*)(ws + 132608);        // 262144 B
  unsigned char* bp        = (unsigned char*)(ws + 394752);  // 16 MB

  float* out      = (float*)d_out;
  float* out_loss = out;
  float* out_pred = out + 1;
  float* out_lp   = out + 1 + (size_t)Bn * Tn;

  prep_rfrag_kernel<<<(Kn * Kn) / 256, 256, 0, stream>>>(trans, Rfrag);
  score_kernel<<<Bn, 32, 0, stream>>>(logit, trans, startT, endT, target, seq_lens, score);
  fwd_kernel<<<Bn / 16, 512, 0, stream>>>(logit, startT, endT, seq_lens, Rfrag, logZ);
  loss_kernel<<<1, 32, 0, stream>>>(logZ, score, out_loss);
  viterbi_kernel<<<Bn, 256, 0, stream>>>(logit, trans, startT, endT, seq_lens, bp, best_last);
  backtrack_kernel<<<1, 128, 0, stream>>>(bp, best_last, pred_i, out_pred);
  out_kernel<<<(Bn * Tn) / 8, 256, 0, stream>>>(logit, pred_i, out_lp);
}
